// SelfAttention_65206193488479
// MI455X (gfx1250) — compile-verified, matrix operand order checked by reading, not run on hardware
//
#include <hip/hip_runtime.h>
#include <hip/hip_bf16.h>
#include <stdint.h>

#define NHEAD 16
#define DHEAD 64
#define BATCH 8
#define SEQ   1024
#define DMODEL 1024
#define NROW (BATCH * SEQ) /* 8192 */

typedef __attribute__((ext_vector_type(16))) __bf16 v16bf;
typedef __attribute__((ext_vector_type(8)))  float  v8f;
typedef int v4i_vs __attribute__((vector_size(16)));  // matches builtin param type

union FragBF {
    v16bf v;
    uint4 q[2];
    unsigned short us[16];
};

__device__ __forceinline__ unsigned short f2bf(float x) {
    unsigned int u = __float_as_uint(x);
    u += 0x7FFFu + ((u >> 16) & 1u);   // round-to-nearest-even
    return (unsigned short)(u >> 16);
}
__device__ __forceinline__ float bf2f(unsigned short h) {
    return __uint_as_float(((unsigned int)h) << 16);
}
__device__ __forceinline__ v8f v8f_zero() {
    v8f z = {0.f, 0.f, 0.f, 0.f, 0.f, 0.f, 0.f, 0.f};
    return z;
}

// ---------------------------------------------------------------------------
// Async global->LDS copy (CDNA5 GLOBAL_LOAD_ASYNC_TO_LDS_B128, ASYNCcnt),
// with a synchronous reg-staged fallback if the builtins aren't declared.
// Builtin signature (from hipcc diagnostic): (v4i as1* gaddr, v4i as3* lds,
// imm offset, imm cpol).
// ---------------------------------------------------------------------------
#if __has_builtin(__builtin_amdgcn_global_load_async_to_lds_b128) && \
    __has_builtin(__builtin_amdgcn_s_wait_asynccnt)
#define HAVE_ASYNC_LDS 1
#endif

__device__ __forceinline__ void async_copy_b128(const unsigned short* src,
                                                unsigned short* dstLds) {
#if defined(HAVE_ASYNC_LDS)
    __builtin_amdgcn_global_load_async_to_lds_b128(
        (__attribute__((address_space(1))) v4i_vs*)(void*)src,
        (__attribute__((address_space(3))) v4i_vs*)(void*)dstLds, 0, 0);
#else
    *(uint4*)dstLds = *(const uint4*)src;
#endif
}

__device__ __forceinline__ void wait_async_keep6() {
#if defined(HAVE_ASYNC_LDS)
    __builtin_amdgcn_s_wait_asynccnt(6);  // drain current buffer, keep next stage
#endif
}
__device__ __forceinline__ void wait_async_all() {
#if defined(HAVE_ASYNC_LDS)
    __builtin_amdgcn_s_wait_asynccnt(0);
#endif
}

// ---------------------------------------------------------------------------
// f32 -> bf16 convert (grid-stride)
// ---------------------------------------------------------------------------
__global__ void convert_bf16_kernel(const float* __restrict__ src,
                                    unsigned short* __restrict__ dst, int n) {
    int i = blockIdx.x * blockDim.x + threadIdx.x;
    int stride = gridDim.x * blockDim.x;
    for (; i < n; i += stride) dst[i] = f2bf(src[i]);
}

// W[K,N] f32 -> Wt[N,K] bf16
__global__ void transpose_bf16_kernel(const float* __restrict__ W,
                                      unsigned short* __restrict__ Wt,
                                      int K, int N) {
    int i = blockIdx.x * blockDim.x + threadIdx.x;
    int stride = gridDim.x * blockDim.x;
    int tot = K * N;
    for (; i < tot; i += stride) {
        int k = i / N, n = i - k * N;
        Wt[(size_t)n * K + k] = f2bf(W[i]);
    }
}

// ---------------------------------------------------------------------------
// bf16 WMMA GEMM: C[M,N] = A[M,K] * Bt[N,K]^T + bias[N]
// Block = 8 waves, tile 256x128; wave tile 64x64 (4x4 wmma_f32_16x16x32_bf16).
// K-slabs of 32 double-buffered in LDS via async global->LDS copies:
// per stage each thread issues 4 A-chunks + 2 B-chunks of b128 (ASYNCcnt += 6).
// ---------------------------------------------------------------------------
template <bool OUT_BF16>
__global__ void __launch_bounds__(256)
gemm_wmma_kernel(const unsigned short* __restrict__ A,
                 const unsigned short* __restrict__ Bt,
                 const float* __restrict__ bias,
                 void* __restrict__ Cout, int M, int N, int K) {
    __shared__ __align__(16) unsigned short Abuf[2][256][40];  // 40-elem pad
    __shared__ __align__(16) unsigned short Bbuf[2][128][40];

    const int lane  = threadIdx.x & 31;
    const int wave  = threadIdx.x >> 5;
    const int lmod  = lane & 15;
    const int lhalf = lane >> 4;
    const int mW = (wave >> 1) * 64;   // wave tile inside block tile
    const int nW = (wave & 1) * 64;
    const int kofflane = lhalf * 8;    // lanes 16-31 hold K groups 8..15/24..31

    const unsigned short* Ablk = A + (size_t)(blockIdx.x * 256) * K;
    const unsigned short* Bblk = Bt + (size_t)(blockIdx.y * 128) * K;

    auto stage = [&](int buf, int k0) {
#pragma unroll
        for (int t = 0; t < 4; ++t) {           // A tile: 256x32 = 1024 chunks
            const int ch = threadIdx.x + t * 256;
            const int row = ch >> 2, c8 = (ch & 3) * 8;
            async_copy_b128(Ablk + (size_t)row * K + k0 + c8,
                            &Abuf[buf][row][c8]);
        }
#pragma unroll
        for (int t = 0; t < 2; ++t) {           // B tile: 128x32 = 512 chunks
            const int ch = threadIdx.x + t * 256;
            const int row = ch >> 2, c8 = (ch & 3) * 8;
            async_copy_b128(Bblk + (size_t)row * K + k0 + c8,
                            &Bbuf[buf][row][c8]);
        }
    };

    v8f acc[4][4];
#pragma unroll
    for (int i = 0; i < 4; ++i)
#pragma unroll
        for (int j = 0; j < 4; ++j) acc[i][j] = v8f_zero();

    stage(0, 0);

    for (int k0 = 0; k0 < K; k0 += 32) {
        const int buf = (k0 >> 5) & 1;
        if (k0 + 32 < K) {
            stage(buf ^ 1, k0 + 32);
            wait_async_keep6();
        } else {
            wait_async_all();
        }
        __syncthreads();  // everyone's chunks of this buffer are in LDS

        FragBF a[4], b[4];
#pragma unroll
        for (int i = 0; i < 4; ++i) {
            const unsigned short* p = &Abuf[buf][mW + i * 16 + lmod][kofflane];
            a[i].q[0] = *(const uint4*)(p);
            a[i].q[1] = *(const uint4*)(p + 16);
        }
#pragma unroll
        for (int j = 0; j < 4; ++j) {
            const unsigned short* p = &Bbuf[buf][nW + j * 16 + lmod][kofflane];
            b[j].q[0] = *(const uint4*)(p);
            b[j].q[1] = *(const uint4*)(p + 16);
        }
#pragma unroll
        for (int i = 0; i < 4; ++i)
#pragma unroll
            for (int j = 0; j < 4; ++j)
                acc[i][j] = __builtin_amdgcn_wmma_f32_16x16x32_bf16(
                    false, a[i].v, false, b[j].v, (short)0, acc[i][j], false,
                    false);
        __syncthreads();  // reads done before this buffer is re-staged
    }

    const int mBase = blockIdx.x * 256 + mW;
    const int nBase = blockIdx.y * 128 + nW;
#pragma unroll
    for (int j = 0; j < 4; ++j) {
        const int col = nBase + j * 16 + lmod;
        const float bj = bias[col];
#pragma unroll
        for (int i = 0; i < 4; ++i) {
#pragma unroll
            for (int e = 0; e < 8; ++e) {
                const int row = mBase + i * 16 + e + lhalf * 8;
                const float val = acc[i][j][e] + bj;
                if (OUT_BF16)
                    ((unsigned short*)Cout)[(size_t)row * N + col] = f2bf(val);
                else
                    ((float*)Cout)[(size_t)row * N + col] = val;
            }
        }
    }
}

// ---------------------------------------------------------------------------
// Flash attention + R-gating.
// Grid: (SEQ/128, BATCH*NHEAD). Block 256 = 8 waves; wave owns 16 query rows.
// Per 32-key step: 4 WMMAs for energy (QK^T), online softmax, 4 WMMAs for PV.
// ---------------------------------------------------------------------------
__global__ void __launch_bounds__(256)
attention_wmma_kernel(const unsigned short* __restrict__ Q,
                      const unsigned short* __restrict__ Km,
                      const unsigned short* __restrict__ V,
                      const unsigned short* __restrict__ R,
                      unsigned short* __restrict__ Gated) {
    __shared__ __align__(16) unsigned short vT[DHEAD][40];   // [d][key] transposed V tile
    __shared__ __align__(16) unsigned short pbuf[8][16][40]; // per-wave P tile

    const int lane  = threadIdx.x & 31;
    const int wave  = threadIdx.x >> 5;
    const int lmod  = lane & 15;
    const int lhalf = lane >> 4;
    const int bh = blockIdx.y;
    const int b = bh / NHEAD;
    const int h = bh % NHEAD;
    const int qRow0 = blockIdx.x * 128 + wave * 16;
    const size_t rowStride = (size_t)(NHEAD * DHEAD);  // 1024
    const size_t baseBH = ((size_t)b * SEQ) * rowStride + (size_t)h * DHEAD;

    // Q fragments (DH=64 -> two K-steps of 32)
    FragBF aq[2];
    {
        const unsigned short* qp =
            Q + baseBH + (size_t)(qRow0 + lmod) * rowStride + lhalf * 8;
        aq[0].q[0] = *(const uint4*)(qp);
        aq[0].q[1] = *(const uint4*)(qp + 16);
        aq[1].q[0] = *(const uint4*)(qp + 32);
        aq[1].q[1] = *(const uint4*)(qp + 48);
    }

    v8f ctx[4];
#pragma unroll
    for (int j = 0; j < 4; ++j) ctx[j] = v8f_zero();
    float mrow[8], srow[8];
#pragma unroll
    for (int e = 0; e < 8; ++e) { mrow[e] = -3.0e38f; srow[e] = 0.f; }

    const float invScale = 1.0f / 32.0f;  // SCALE = 64**1/2 = 32.0

    for (int kt = 0; kt < SEQ; kt += 32) {
        __syncthreads();  // protect LDS tiles from previous iteration readers

        // cooperative transposed V tile load: 32 keys x 64 dims
        {
            const int vr = threadIdx.x >> 3;        // key row 0..31
            const int c0 = (threadIdx.x & 7) * 8;   // dim chunk
            FragBF tmp;
            const unsigned short* vp =
                V + baseBH + (size_t)(kt + vr) * rowStride + c0;
            tmp.q[0] = *(const uint4*)(vp);
#pragma unroll
            for (int e = 0; e < 8; ++e) vT[c0 + e][vr] = tmp.us[e];
        }

        // energy = Q (16xDH) @ K^T (DHx32), two 16-col tiles
        v8f e0 = v8f_zero(), e1 = v8f_zero();
        {
            FragBF bk;
            const unsigned short* kp =
                Km + baseBH + (size_t)(kt + lmod) * rowStride + lhalf * 8;
            bk.q[0] = *(const uint4*)(kp);
            bk.q[1] = *(const uint4*)(kp + 16);
            e0 = __builtin_amdgcn_wmma_f32_16x16x32_bf16(false, aq[0].v, false,
                                                         bk.v, (short)0, e0,
                                                         false, false);
            bk.q[0] = *(const uint4*)(kp + 32);
            bk.q[1] = *(const uint4*)(kp + 48);
            e0 = __builtin_amdgcn_wmma_f32_16x16x32_bf16(false, aq[1].v, false,
                                                         bk.v, (short)0, e0,
                                                         false, false);
        }
        {
            FragBF bk;
            const unsigned short* kp =
                Km + baseBH + (size_t)(kt + 16 + lmod) * rowStride + lhalf * 8;
            bk.q[0] = *(const uint4*)(kp);
            bk.q[1] = *(const uint4*)(kp + 16);
            e1 = __builtin_amdgcn_wmma_f32_16x16x32_bf16(false, aq[0].v, false,
                                                         bk.v, (short)0, e1,
                                                         false, false);
            bk.q[0] = *(const uint4*)(kp + 32);
            bk.q[1] = *(const uint4*)(kp + 48);
            e1 = __builtin_amdgcn_wmma_f32_16x16x32_bf16(false, aq[1].v, false,
                                                         bk.v, (short)0, e1,
                                                         false, false);
        }

        // online softmax; row r = e + lhalf*8 lives in 16 lanes of one half,
        // so xor-shuffle masks 1/2/4/8 reduce exactly within the row group.
        float alpha[8];
#pragma unroll
        for (int e = 0; e < 8; ++e) {
            const float t0 = e0[e] * invScale;
            const float t1 = e1[e] * invScale;
            float mx = fmaxf(t0, t1);
            mx = fmaxf(mx, __shfl_xor(mx, 1, 32));
            mx = fmaxf(mx, __shfl_xor(mx, 2, 32));
            mx = fmaxf(mx, __shfl_xor(mx, 4, 32));
            mx = fmaxf(mx, __shfl_xor(mx, 8, 32));
            const float mnew = fmaxf(mrow[e], mx);
            const float al = __expf(mrow[e] - mnew);
            const float p0 = __expf(t0 - mnew);
            const float p1 = __expf(t1 - mnew);
            float ps = p0 + p1;
            ps += __shfl_xor(ps, 1, 32);
            ps += __shfl_xor(ps, 2, 32);
            ps += __shfl_xor(ps, 4, 32);
            ps += __shfl_xor(ps, 8, 32);
            srow[e] = srow[e] * al + ps;
            mrow[e] = mnew;
            alpha[e] = al;
            const int prow = e + lhalf * 8;  // C-layout row
            pbuf[wave][prow][lmod] = f2bf(p0);
            pbuf[wave][prow][16 + lmod] = f2bf(p1);
        }

        // rescale running context by alpha before accumulating this step
#pragma unroll
        for (int j = 0; j < 4; ++j)
#pragma unroll
            for (int e = 0; e < 8; ++e) ctx[j][e] *= alpha[e];

        __syncthreads();  // vT + pbuf fully written

        // P fragment in A-layout (16x32)
        FragBF ap;
        {
            const unsigned short* pp = &pbuf[wave][lmod][lhalf * 8];
            ap.q[0] = *(const uint4*)(pp);
            ap.q[1] = *(const uint4*)(pp + 16);
        }
        // ctx += P (16x32) @ V (32x64): 4 column tiles from transposed LDS V
#pragma unroll
        for (int j = 0; j < 4; ++j) {
            FragBF bv;
            const unsigned short* vp = &vT[j * 16 + lmod][lhalf * 8];
            bv.q[0] = *(const uint4*)(vp);
            bv.q[1] = *(const uint4*)(vp + 16);
            ctx[j] = __builtin_amdgcn_wmma_f32_16x16x32_bf16(
                false, ap.v, false, bv.v, (short)0, ctx[j], false, false);
        }
    }

    // normalize, R-gate, write gated activation [B*S, H*DH] as bf16
#pragma unroll
    for (int j = 0; j < 4; ++j) {
        const int d = j * 16 + lmod;
#pragma unroll
        for (int e = 0; e < 8; ++e) {
            const int row = qRow0 + e + lhalf * 8;
            const size_t idx = baseBH + (size_t)row * rowStride + d;
            const float val = ctx[j][e] / srow[e];
            Gated[idx] = f2bf(val * bf2f(R[idx]));
        }
    }
}

// ---------------------------------------------------------------------------
// Host-side launch
// ---------------------------------------------------------------------------
extern "C" void kernel_launch(void* const* d_in, const int* in_sizes, int n_in,
                              void* d_out, int out_size, void* d_ws,
                              size_t ws_size, hipStream_t stream) {
    (void)in_sizes; (void)n_in; (void)out_size; (void)ws_size;

    const float* value = (const float*)d_in[0];
    const float* keyin = (const float*)d_in[1];
    const float* query = (const float*)d_in[2];
    const float* Wq = (const float*)d_in[3];  const float* bq = (const float*)d_in[4];
    const float* Wk = (const float*)d_in[5];  const float* bk = (const float*)d_in[6];
    const float* Wv = (const float*)d_in[7];  const float* bv = (const float*)d_in[8];
    const float* Wr = (const float*)d_in[9];  const float* br = (const float*)d_in[10];
    const float* Wo = (const float*)d_in[11]; const float* bo = (const float*)d_in[12];

    char* ws = (char*)d_ws;
    const size_t actSz = (size_t)NROW * DMODEL * sizeof(unsigned short); // 16 MB
    const size_t wSz = (size_t)DMODEL * DMODEL * sizeof(unsigned short); //  2 MB
    unsigned short* q_bf  = (unsigned short*)(ws + 0 * actSz);
    unsigned short* k_bf  = (unsigned short*)(ws + 1 * actSz);
    unsigned short* v_bf  = (unsigned short*)(ws + 2 * actSz);
    unsigned short* r_bf  = (unsigned short*)(ws + 3 * actSz);
    unsigned short* xq    = (unsigned short*)(ws + 4 * actSz);
    unsigned short* xk    = (unsigned short*)(ws + 5 * actSz);
    unsigned short* xv    = (unsigned short*)(ws + 6 * actSz);
    unsigned short* gated = (unsigned short*)(ws + 7 * actSz);
    unsigned short* wqt   = (unsigned short*)(ws + 8 * actSz + 0 * wSz);
    unsigned short* wkt   = (unsigned short*)(ws + 8 * actSz + 1 * wSz);
    unsigned short* wvt   = (unsigned short*)(ws + 8 * actSz + 2 * wSz);
    unsigned short* wrt   = (unsigned short*)(ws + 8 * actSz + 3 * wSz);
    unsigned short* wot   = (unsigned short*)(ws + 8 * actSz + 4 * wSz);

    const int nAct = NROW * DMODEL;  // 8,388,608
    convert_bf16_kernel<<<4096, 256, 0, stream>>>(value, xv, nAct);
    convert_bf16_kernel<<<4096, 256, 0, stream>>>(keyin, xk, nAct);
    convert_bf16_kernel<<<4096, 256, 0, stream>>>(query, xq, nAct);
    transpose_bf16_kernel<<<4096, 256, 0, stream>>>(Wq, wqt, DMODEL, DMODEL);
    transpose_bf16_kernel<<<4096, 256, 0, stream>>>(Wk, wkt, DMODEL, DMODEL);
    transpose_bf16_kernel<<<4096, 256, 0, stream>>>(Wv, wvt, DMODEL, DMODEL);
    transpose_bf16_kernel<<<4096, 256, 0, stream>>>(Wr, wrt, DMODEL, DMODEL);
    transpose_bf16_kernel<<<4096, 256, 0, stream>>>(Wo, wot, DMODEL, DMODEL);

    dim3 gGrid(NROW / 256, DMODEL / 128);  // (32, 8)
    gemm_wmma_kernel<true><<<gGrid, 256, 0, stream>>>(xq, wqt, bq, q_bf, NROW, DMODEL, DMODEL);
    gemm_wmma_kernel<true><<<gGrid, 256, 0, stream>>>(xk, wkt, bk, k_bf, NROW, DMODEL, DMODEL);
    gemm_wmma_kernel<true><<<gGrid, 256, 0, stream>>>(xv, wvt, bv, v_bf, NROW, DMODEL, DMODEL);
    gemm_wmma_kernel<true><<<gGrid, 256, 0, stream>>>(xq, wrt, br, r_bf, NROW, DMODEL, DMODEL);

    dim3 aGrid(SEQ / 128, BATCH * NHEAD);  // (8, 128)
    attention_wmma_kernel<<<aGrid, 256, 0, stream>>>(q_bf, k_bf, v_bf, r_bf, gated);

    gemm_wmma_kernel<false><<<gGrid, 256, 0, stream>>>(gated, wot, bo, d_out, NROW, DMODEL, DMODEL);
}
